// tagger_7249904795857
// MI455X (gfx1250) — compile-verified, matrix operand order checked by reading, not run on hardware
//
#include <hip/hip_runtime.h>
#include <hip/hip_bf16.h>

// ---------------------------------------------------------------------------
// GRU tagger for MI455X (gfx1250, wave32, WMMA).
//   T=2048, E=1024, H=1024, 3H=3072, TAGS=50
// Pipeline (all on `stream`):
//   1. init (zero h ping-pong buffers + sync counter)
//   2. embedding gather -> x_bf16
//   3. f32->bf16 weight conversion (W_ih, W_hh, W_out padded to 64 rows)
//   4. gi = x @ W_ih^T + b_ih      (WMMA bf16, f32 accum)  [2048,3072]
//   5. persistent GRU scan: 32 WGs, W_hh slice resident in LDS (192KB/WGP,
//      async-filled via GLOBAL_LOAD_ASYNC_TO_LDS_B128), WMMA GEMV per step,
//      device-wide barrier per step
//   6. tags = hs @ W_out^T + b_out, fused log_softmax      (WMMA)
// ---------------------------------------------------------------------------

#define T_LEN 2048
#define E_DIM 1024
#define H_DIM 1024
#define G3    (3 * H_DIM)
#define NTAGS 50
#define TAGPAD 64

#define SCAN_WGS 32
#define SCAN_THREADS 192   // 6 waves: 3 gates x 2 waves x 16 rows = 96 rows
#define SCAN_ROWS 96       // W_hh rows per WG
#define ROWP (H_DIM + 8)   // LDS row stride (bf16): 2064B -> conflict-free b128

typedef __bf16 bf16_t;
typedef bf16_t v16bf __attribute__((ext_vector_type(16)));
typedef bf16_t bf8v  __attribute__((ext_vector_type(8)));
typedef float  v8f   __attribute__((ext_vector_type(8)));
typedef int    v4i   __attribute__((ext_vector_type(4)));
typedef __attribute__((address_space(1))) v4i as1_v4i;
typedef __attribute__((address_space(3))) v4i as3_v4i;

#if __has_builtin(__builtin_amdgcn_global_load_async_to_lds_b128) && \
    __has_builtin(__builtin_amdgcn_s_wait_asynccnt)
#define USE_ASYNC_LDS 1
#endif

// ---- WMMA helpers ----------------------------------------------------------

static __device__ __forceinline__ v8f wmma_bf16(v16bf a, v16bf b, v8f c) {
  return __builtin_amdgcn_wmma_f32_16x16x32_bf16(
      /*neg_a=*/false, a, /*neg_b=*/false, b,
      /*c_mod=*/(short)0, c, /*reuse_a=*/false, /*reuse_b=*/false);
}

// Load a 16x32 bf16 fragment from a row-major matrix (K contiguous).
// ISA 7.12.2: lane&15 = row; lanes 0-15 hold K {0..7,16..23}, lanes 16-31
// hold K {8..15,24..31}; each half is one contiguous 16B chunk.
static __device__ __forceinline__ v16bf load_frag(const bf16_t* __restrict__ base,
                                                  int row_stride) {
  const int lane = threadIdx.x & 31;
  const int r    = lane & 15;
  const int kh   = (lane >> 4) << 3;          // 0 or 8
  const bf16_t* p = base + (size_t)r * row_stride;
  bf8v lo = *reinterpret_cast<const bf8v*>(p + kh);
  bf8v hi = *reinterpret_cast<const bf8v*>(p + 16 + kh);
  v16bf f;
#pragma unroll
  for (int i = 0; i < 8; ++i) { f[i] = lo[i]; f[8 + i] = hi[i]; }
  return f;
}

// Broadcast one 32-long bf16 vector chunk into all 16 A rows (GEMV trick).
static __device__ __forceinline__ v16bf load_frag_bcast(const bf16_t* __restrict__ v) {
  const int lane = threadIdx.x & 31;
  const int kh   = (lane >> 4) << 3;
  bf8v lo = *reinterpret_cast<const bf8v*>(v + kh);
  bf8v hi = *reinterpret_cast<const bf8v*>(v + 16 + kh);
  v16bf f;
#pragma unroll
  for (int i = 0; i < 8; ++i) { f[i] = lo[i]; f[8 + i] = hi[i]; }
  return f;
}

// ---- small prep kernels ----------------------------------------------------

__global__ void k_init(float* hf, bf16_t* hb, unsigned* cnt) {
  int i = blockIdx.x * blockDim.x + threadIdx.x;
  if (i < 2 * H_DIM) { hf[i] = 0.0f; hb[i] = (bf16_t)0.0f; }
  if (i == 0) *cnt = 0u;
}

__global__ void k_embed(const int* __restrict__ sent,
                        const float* __restrict__ emb,
                        bf16_t* __restrict__ xb) {
  const int t   = blockIdx.x;
  const int idx = sent[t];
  const float* src = emb + (size_t)idx * E_DIM;
  bf16_t* dst = xb + (size_t)t * E_DIM;
  for (int i = threadIdx.x; i < E_DIM; i += blockDim.x) dst[i] = (bf16_t)src[i];
}

__global__ void k_cvt(const float* __restrict__ s, bf16_t* __restrict__ d, int n) {
  int i = blockIdx.x * blockDim.x + threadIdx.x;
  if (i < n) d[i] = (bf16_t)s[i];
}

__global__ void k_cvt_wout(const float* __restrict__ s, bf16_t* __restrict__ d) {
  int i = blockIdx.x * blockDim.x + threadIdx.x;
  if (i >= TAGPAD * H_DIM) return;
  int r = i / H_DIM;
  d[i] = (r < NTAGS) ? (bf16_t)s[i] : (bf16_t)0.0f;
}

// ---- GEMM: gi = x @ W_ih^T + b_ih  ([2048,1024] x [3072,1024]^T) -----------
// Block = 8 waves -> 32x64 output tile (2x4 grid of 16x16 WMMA tiles).

__global__ void __launch_bounds__(256)
k_gemm_gi(const bf16_t* __restrict__ X, const bf16_t* __restrict__ W,
          const float* __restrict__ bias, float* __restrict__ out) {
  const int wave = threadIdx.x >> 5;
  const int lane = threadIdx.x & 31;
  const int wm = wave >> 2, wn = wave & 3;
  const int m0 = blockIdx.y * 32 + wm * 16;
  const int n0 = blockIdx.x * 64 + wn * 16;
  const bf16_t* arow = X + (size_t)m0 * E_DIM;
  const bf16_t* brow = W + (size_t)n0 * E_DIM;
  v8f c = {};
  for (int k = 0; k < E_DIM; k += 32) {
    __builtin_prefetch(arow + k + 128, 0, 1);   // global_prefetch_b8
    __builtin_prefetch(brow + k + 128, 0, 1);
    v16bf a = load_frag(arow + k, E_DIM);
    v16bf b = load_frag(brow + k, E_DIM);
    c = wmma_bf16(a, b, c);
  }
  const int n  = lane & 15;
  const int mo = (lane >> 4) * 8;
  const float bz = bias[n0 + n];
#pragma unroll
  for (int r = 0; r < 8; ++r)
    out[(size_t)(m0 + mo + r) * G3 + n0 + n] = c[r] + bz;
}

// ---- Persistent GRU scan ---------------------------------------------------
// 32 WGs; WG w owns h indices [32w,32w+32) and rows {j, H+j, 2H+j} of W_hh
// (96 rows = 6 waves x 16 rows). W_hh slice lives in LDS (padded rows,
// conflict-free ds_load_b128). Per step: WMMA GEMV (A = broadcast h from L2),
// gate math on 32 lanes, ping-pong h, device-wide arrive/spin barrier.

__global__ void __launch_bounds__(SCAN_THREADS)
k_gru_scan(const bf16_t* __restrict__ Whh, const float* __restrict__ b_hh,
           const float* __restrict__ gi, float* __restrict__ hf,
           bf16_t* __restrict__ hb, bf16_t* __restrict__ hs,
           unsigned* __restrict__ cnt) {
  __shared__ bf16_t ldsw[SCAN_ROWS * ROWP];   // 96 x 1032 bf16 = ~194 KB
  __shared__ float  gh[SCAN_ROWS];
  const int jb   = blockIdx.x * 32;
  const int wave = threadIdx.x >> 5;
  const int lane = threadIdx.x & 31;
  const int g    = wave >> 1;            // gate: 0=r 1=z 2=n
  const int sub  = wave & 1;
  const int row0 = g * H_DIM + jb + sub * 16;   // global row base for this wave
  const int lrow0 = g * 32 + sub * 16;          // LDS row base for this wave

  // One-time LDS fill of this WG's 96 W_hh rows (128 x 16B chunks per row).
  for (int idx = threadIdx.x; idx < SCAN_ROWS * 128; idx += SCAN_THREADS) {
    const int row = idx >> 7;
    const int chunk = idx & 127;
    const int gg = row >> 5, rr = row & 31;
    const bf16_t* gsrc = Whh + ((size_t)(gg * H_DIM + jb + rr)) * H_DIM + chunk * 8;
    bf16_t* ldst = &ldsw[row * ROWP + chunk * 8];
#ifdef USE_ASYNC_LDS
    __builtin_amdgcn_global_load_async_to_lds_b128(
        (as1_v4i*)gsrc, (as3_v4i*)ldst, 0, 0);
#else
    *reinterpret_cast<bf8v*>(ldst) = *reinterpret_cast<const bf8v*>(gsrc);
#endif
  }
#ifdef USE_ASYNC_LDS
  __builtin_amdgcn_s_wait_asynccnt(0);
#endif
  __syncthreads();

  const bf16_t* lwrow = &ldsw[lrow0 * ROWP];

  for (int t = 0; t < T_LEN; ++t) {
    const bf16_t* hcur = hb + (size_t)(t & 1) * H_DIM;
    v8f c = {};
    for (int k = 0; k < H_DIM; k += 32) {
      v16bf a = load_frag_bcast(hcur + k);
      v16bf b = load_frag(lwrow + k, ROWP);     // ds_load_b128, conflict-free
      c = wmma_bf16(a, b, c);
    }
    // All A rows identical -> every C row holds gh for column n = lane&15.
    if (lane < 16)
      gh[lrow0 + lane] = c[0] + b_hh[row0 + lane];
    __syncthreads();

    if (threadIdx.x < 32) {
      const int j = jb + threadIdx.x;
      const size_t gbase = (size_t)t * G3;
      float ir = gi[gbase + j];
      float iz = gi[gbase + H_DIM + j];
      float inn = gi[gbase + 2 * H_DIM + j];
      float hr = gh[threadIdx.x];
      float hz = gh[32 + threadIdx.x];
      float hn = gh[64 + threadIdx.x];
      float r = 1.0f / (1.0f + __expf(-(ir + hr)));
      float z = 1.0f / (1.0f + __expf(-(iz + hz)));
      float nv = tanhf(inn + r * hn);
      float hp = hf[(size_t)(t & 1) * H_DIM + j];
      float hnew = (1.0f - z) * nv + z * hp;
      hf[(size_t)((t + 1) & 1) * H_DIM + j] = hnew;
      hb[(size_t)((t + 1) & 1) * H_DIM + j] = (bf16_t)hnew;
      hs[(size_t)t * H_DIM + j] = (bf16_t)hnew;
    }
    __syncthreads();
    if (threadIdx.x == 0) {
      __hip_atomic_fetch_add(cnt, 1u, __ATOMIC_RELEASE, __HIP_MEMORY_SCOPE_AGENT);
      const unsigned target = (unsigned)(SCAN_WGS * (t + 1));
      while (__hip_atomic_load(cnt, __ATOMIC_ACQUIRE, __HIP_MEMORY_SCOPE_AGENT) < target)
        __builtin_amdgcn_s_sleep(2);
    }
    __syncthreads();
  }
}

// ---- Tag projection + fused log_softmax ------------------------------------
// Block = 4 waves, 16 sequence rows; wave wn -> tag columns [16wn,16wn+16).

__global__ void __launch_bounds__(128)
k_tags(const bf16_t* __restrict__ hsb, const bf16_t* __restrict__ Wout,
       const float* __restrict__ bout, float* __restrict__ out) {
  __shared__ float ts[16][TAGPAD];
  const int wave = threadIdx.x >> 5;
  const int lane = threadIdx.x & 31;
  const int t0 = blockIdx.x * 16;
  const int n0 = wave * 16;
  const bf16_t* arow = hsb + (size_t)t0 * H_DIM;
  const bf16_t* brow = Wout + (size_t)n0 * H_DIM;
  v8f c = {};
  for (int k = 0; k < H_DIM; k += 32) {
    v16bf a = load_frag(arow + k, H_DIM);
    v16bf b = load_frag(brow + k, H_DIM);
    c = wmma_bf16(a, b, c);
  }
  const int n  = lane & 15;
  const int mo = (lane >> 4) * 8;
  const float bz = (n0 + n) < NTAGS ? bout[n0 + n] : 0.0f;
#pragma unroll
  for (int r = 0; r < 8; ++r) ts[mo + r][n0 + n] = c[r] + bz;
  __syncthreads();

  if (threadIdx.x < 16) {
    const int m = threadIdx.x;
    float mx = -1e30f;
    for (int i = 0; i < NTAGS; ++i) mx = fmaxf(mx, ts[m][i]);
    float s = 0.0f;
    for (int i = 0; i < NTAGS; ++i) s += __expf(ts[m][i] - mx);
    const float lse = mx + __logf(s);
    for (int i = 0; i < NTAGS; ++i)
      out[(size_t)(t0 + m) * NTAGS + i] = ts[m][i] - lse;
  }
}

// ---- host launcher ---------------------------------------------------------

extern "C" void kernel_launch(void* const* d_in, const int* in_sizes, int n_in,
                              void* d_out, int out_size, void* d_ws, size_t ws_size,
                              hipStream_t stream) {
  (void)in_sizes; (void)n_in; (void)out_size; (void)ws_size;
  const int*   sentence = (const int*)d_in[0];
  const float* emb      = (const float*)d_in[1];
  const float* W_ih     = (const float*)d_in[2];
  const float* W_hh     = (const float*)d_in[3];
  const float* b_ih     = (const float*)d_in[4];
  const float* b_hh     = (const float*)d_in[5];
  const float* W_out    = (const float*)d_in[6];
  const float* b_out    = (const float*)d_in[7];
  float* out = (float*)d_out;

  char* ws = (char*)d_ws;
  size_t off = 0;
  auto alloc = [&](size_t bytes) -> void* {
    void* p = ws + off;
    off = (off + bytes + 255) & ~(size_t)255;
    return p;
  };
  bf16_t*   xb    = (bf16_t*)alloc((size_t)T_LEN * E_DIM * sizeof(bf16_t));
  bf16_t*   wihb  = (bf16_t*)alloc((size_t)G3 * E_DIM * sizeof(bf16_t));
  bf16_t*   whhb  = (bf16_t*)alloc((size_t)G3 * H_DIM * sizeof(bf16_t));
  bf16_t*   woutb = (bf16_t*)alloc((size_t)TAGPAD * H_DIM * sizeof(bf16_t));
  float*    gi    = (float*) alloc((size_t)T_LEN * G3 * sizeof(float));
  bf16_t*   hsb   = (bf16_t*)alloc((size_t)T_LEN * H_DIM * sizeof(bf16_t));
  float*    hf    = (float*) alloc(2 * H_DIM * sizeof(float));
  bf16_t*   hb    = (bf16_t*)alloc(2 * H_DIM * sizeof(bf16_t));
  unsigned* cnt   = (unsigned*)alloc(256);

  k_init<<<8, 256, 0, stream>>>(hf, hb, cnt);
  k_embed<<<T_LEN, 256, 0, stream>>>(sentence, emb, xb);
  k_cvt<<<(G3 * E_DIM + 255) / 256, 256, 0, stream>>>(W_ih, wihb, G3 * E_DIM);
  k_cvt<<<(G3 * H_DIM + 255) / 256, 256, 0, stream>>>(W_hh, whhb, G3 * H_DIM);
  k_cvt_wout<<<(TAGPAD * H_DIM + 255) / 256, 256, 0, stream>>>(W_out, woutb);

  dim3 gg(G3 / 64, T_LEN / 32);
  k_gemm_gi<<<gg, 256, 0, stream>>>(xb, wihb, b_ih, gi);

  k_gru_scan<<<SCAN_WGS, SCAN_THREADS, 0, stream>>>(whhb, b_hh, gi, hf, hb, hsb, cnt);

  k_tags<<<T_LEN / 16, 128, 0, stream>>>(hsb, woutb, b_out, out);
}
